// Qwen3Attention_81295140978789
// MI455X (gfx1250) — matrix operand order repaired and checked by reading, not compiled, not run on hardware
//
#include <hip/hip_runtime.h>
#include <hip/hip_bf16.h>

// Problem constants (match reference)
#define B_    2
#define S_    2048
#define HID_  2048
#define NH_   32
#define NKV_  8
#define D_    128
#define GRP_  (NH_ / NKV_)
#define SCALE_ 0.08838834764831845f   // 1/sqrt(128)

typedef __attribute__((ext_vector_type(16))) __bf16 v16bf;
typedef __attribute__((ext_vector_type(8)))  float  v8f;

// CDNA5 feature probes (device pass only); fall back to portable paths.
#if defined(__HIP_DEVICE_COMPILE__) && \
    __has_builtin(__builtin_amdgcn_global_load_async_to_lds_b128) && \
    __has_builtin(__builtin_amdgcn_s_wait_asynccnt)
#define USE_ASYNC_LDS 1
#else
#define USE_ASYNC_LDS 0
#endif

#if defined(__HIP_DEVICE_COMPILE__) && \
    __has_builtin(__builtin_amdgcn_tensor_load_to_lds) && \
    __has_builtin(__builtin_amdgcn_s_wait_tensorcnt)
#define USE_TDM 1
#else
#define USE_TDM 0
#endif

#if USE_ASYNC_LDS || USE_TDM
__device__ __forceinline__ unsigned lds_offset(const void* p) {
  return (unsigned)(size_t)(const __attribute__((address_space(3))) void*)p;
}
#endif

#if USE_ASYNC_LDS
// Builtin params are non-const pointers to 16-byte int vectors in AS(1)/AS(3)
// (clang prints AS(1) as "__device__" in HIP mode).
typedef int v4i_gcc __attribute__((vector_size(16)));
__device__ __forceinline__ void async_b128(const void* g, void* l) {
  __builtin_amdgcn_global_load_async_to_lds_b128(
      (__attribute__((address_space(1))) v4i_gcc*)g,
      (__attribute__((address_space(3))) v4i_gcc*)l, 0, 0);
}
#endif

#if USE_TDM
typedef unsigned int u32x4 __attribute__((ext_vector_type(4)));
typedef int          i32x8 __attribute__((ext_vector_type(8)));
typedef int          i32x4 __attribute__((ext_vector_type(4)));
#endif

union Frag16 {
  v16bf v;
  unsigned int d[8];
};

__device__ __forceinline__ unsigned short f32_to_bf16_u(float f) {
  unsigned int u = __float_as_uint(f);
  u += 0x7fffu + ((u >> 16) & 1u);   // round-to-nearest-even
  return (unsigned short)(u >> 16);
}

// A-matrix fragment (16x32 bf16, MxK), per CDNA5 ISA 7.12.2:
// lane L holds row M = L&15; lanes>=16 use K-base 8;
// VGPR v in 0..3 -> K = kb+2v,kb+2v+1 ; v in 4..7 -> K = 16+kb+2(v-4),+1
__device__ __forceinline__ v16bf frag_ld_a(const unsigned short* base, int row,
                                           int ld, int lane) {
  const unsigned short* p = base + row * ld + ((lane & 16) ? 8 : 0);
  Frag16 f;
#pragma unroll
  for (int v = 0; v < 4; ++v) {
    f.d[v]     = *(const unsigned int*)(p + 2 * v);
    f.d[v + 4] = *(const unsigned int*)(p + 16 + 2 * v);
  }
  return f.v;
}

// B-matrix fragment (32x16 bf16, KxN), per ISA B layout:
// lane L holds column N = L&15; same K order as A: lanes>=16 use K-base 8,
// VGPR v in 0..3 -> K = kb+2v,+1 ; v in 4..7 -> K = 16+kb+2(v-4),+1
// Storage assumed [N][K] row-major (row = n, col = k).
__device__ __forceinline__ v16bf frag_ld_b(const unsigned short* base, int row,
                                           int ld, int lane) {
  const unsigned short* p = base + row * ld + ((lane & 16) ? 8 : 0);
  Frag16 f;
#pragma unroll
  for (int v = 0; v < 4; ++v) {
    f.d[v]     = *(const unsigned int*)(p + 2 * v);
    f.d[v + 4] = *(const unsigned int*)(p + 16 + 2 * v);
  }
  return f.v;
}

// ---------------------------------------------------------------------------
// fp32 -> bf16 elementwise conversion (4 elems/thread)
// ---------------------------------------------------------------------------
__global__ __launch_bounds__(256) void cvt_bf16_kernel(
    const float* __restrict__ in, unsigned short* __restrict__ out, int n) {
  int i = (blockIdx.x * 256 + threadIdx.x) * 4;
  if (i >= n) return;
  float4 v = *(const float4*)(in + i);
  unsigned int lo = (unsigned int)f32_to_bf16_u(v.x) |
                    ((unsigned int)f32_to_bf16_u(v.y) << 16);
  unsigned int hi = (unsigned int)f32_to_bf16_u(v.z) |
                    ((unsigned int)f32_to_bf16_u(v.w) << 16);
  *(uint2*)(out + i) = make_uint2(lo, hi);
}

// ---------------------------------------------------------------------------
// GEMM: C[M,N] = A[M,K](bf16) @ W[N,K](bf16)^T, fp32 accumulate.
// Block tile 128x128, 8 waves (4x2), wave tile 32x64, K-step 32.
// Async path: double-buffered LDS, GLOBAL_LOAD_ASYNC_TO_LDS_B128 staging of
// the next K tile overlapped with WMMA on the current one.
// Requires M%128==0, N%128==0, K%32==0 (true for all calls here).
// ---------------------------------------------------------------------------
__global__ __launch_bounds__(256) void gemm_bf16_kernel(
    const unsigned short* __restrict__ A, const unsigned short* __restrict__ W,
    void* __restrict__ C, int M, int N, int K, int out_bf16) {
#if USE_ASYNC_LDS
  __shared__ unsigned short As[2][128 * 40];  // 32 cols + 8 pad
  __shared__ unsigned short Ws[2][128 * 40];
#else
  __shared__ unsigned short As[1][128 * 40];
  __shared__ unsigned short Ws[1][128 * 40];
#endif
  const int tid  = threadIdx.x;
  const int lane = tid & 31;
  const int wave = tid >> 5;
  const int wm = wave >> 1;   // 0..3
  const int wn = wave & 1;    // 0..1
  const int m0 = blockIdx.y * 128;
  const int n0 = blockIdx.x * 128;

  v8f zero = {};
  v8f acc[2][4];
#pragma unroll
  for (int i = 0; i < 2; ++i)
#pragma unroll
    for (int j = 0; j < 4; ++j) acc[i][j] = zero;

  auto compute_tile = [&](const unsigned short* Asb,
                          const unsigned short* Wsb) {
    v16bf af[2], wf[4];
#pragma unroll
    for (int ms = 0; ms < 2; ++ms)
      af[ms] = frag_ld_a(Asb, wm * 32 + ms * 16 + (lane & 15), 40, lane);
#pragma unroll
    for (int ns = 0; ns < 4; ++ns)
      wf[ns] = frag_ld_b(Wsb, wn * 64 + ns * 16 + (lane & 15), 40, lane);
#pragma unroll
    for (int ms = 0; ms < 2; ++ms)
#pragma unroll
      for (int ns = 0; ns < 4; ++ns)
        acc[ms][ns] = __builtin_amdgcn_wmma_f32_16x16x32_bf16(
            false, af[ms], false, wf[ns], (short)0, acc[ms][ns], false, false);
  };

#if USE_ASYNC_LDS
  auto stage = [&](int buf, int k0) {
#pragma unroll
    for (int i = 0; i < 2; ++i) {
      int idx = tid + i * 256;   // 512 chunks: 128 rows x 4 x b128
      int row = idx >> 2;
      int cc  = idx & 3;
      async_b128(A + (size_t)(m0 + row) * K + k0 + cc * 8,
                 &As[buf][row * 40 + cc * 8]);
      async_b128(W + (size_t)(n0 + row) * K + k0 + cc * 8,
                 &Ws[buf][row * 40 + cc * 8]);
    }
  };
  stage(0, 0);
  __builtin_amdgcn_s_wait_asynccnt(0);
  __syncthreads();
  for (int k0 = 0; k0 < K; k0 += 32) {
    const int buf = (k0 >> 5) & 1;
    if (k0 + 32 < K) stage(buf ^ 1, k0 + 32);  // overlap copy with compute
    compute_tile(As[buf], Ws[buf]);
    __builtin_amdgcn_s_wait_asynccnt(0);
    __syncthreads();
  }
#else
  for (int k0 = 0; k0 < K; k0 += 32) {
    uint4 ta[2], tw[2];
#pragma unroll
    for (int i = 0; i < 2; ++i) {   // issue all global loads first
      int idx = tid + i * 256;
      int row = idx >> 2;
      int cc  = idx & 3;
      ta[i] = *(const uint4*)(A + (size_t)(m0 + row) * K + k0 + cc * 8);
      tw[i] = *(const uint4*)(W + (size_t)(n0 + row) * K + k0 + cc * 8);
      if (k0 + 32 < K) {
        __builtin_prefetch(A + (size_t)(m0 + row) * K + k0 + 32 + cc * 8, 0, 0);
        __builtin_prefetch(W + (size_t)(n0 + row) * K + k0 + 32 + cc * 8, 0, 0);
      }
    }
#pragma unroll
    for (int i = 0; i < 2; ++i) {   // then all LDS stores
      int idx = tid + i * 256;
      int row = idx >> 2;
      int cc  = idx & 3;
      *(uint4*)(&As[0][row * 40 + cc * 8]) = ta[i];
      *(uint4*)(&Ws[0][row * 40 + cc * 8]) = tw[i];
    }
    __syncthreads();
    compute_tile(As[0], Ws[0]);
    __syncthreads();
  }
#endif

  const int half = lane >> 4;
  const int nn = lane & 15;
#pragma unroll
  for (int ms = 0; ms < 2; ++ms)
#pragma unroll
    for (int ns = 0; ns < 4; ++ns)
#pragma unroll
      for (int r = 0; r < 8; ++r) {
        int m = m0 + wm * 32 + ms * 16 + r + 8 * half;
        int n = n0 + wn * 64 + ns * 16 + nn;
        float val = acc[ms][ns][r];
        if (out_bf16)
          ((unsigned short*)C)[(size_t)m * N + n] = f32_to_bf16_u(val);
        else
          ((float*)C)[(size_t)m * N + n] = val;
      }
}

// ---------------------------------------------------------------------------
// Per-head RMSNorm + RoPE + relayout [B*S, heads, D](f32) -> [B,heads,S,D](bf16)
// One wave32 per (token, head); lane owns d, d+32, d+64, d+96.
// do_rope==0: plain convert+relayout (for V).
// ---------------------------------------------------------------------------
__global__ __launch_bounds__(256) void normrope_kernel(
    const float* __restrict__ in, const float* __restrict__ nw,
    const float* __restrict__ cosT, const float* __restrict__ sinT,
    unsigned short* __restrict__ out, int heads, int do_rope) {
  const int lane = threadIdx.x & 31;
  const int wv = threadIdx.x >> 5;
  const long item = (long)blockIdx.x * 8 + wv;
  const int head = (int)(item % heads);
  const long token = item / heads;
  const int b = (int)(token / S_);
  const int s = (int)(token % S_);

  const float* x = in + (size_t)token * heads * D_ + (size_t)head * D_;
  float x0 = x[lane], x1 = x[lane + 32], x2 = x[lane + 64], x3 = x[lane + 96];

  if (do_rope) {
    float ss = x0 * x0 + x1 * x1 + x2 * x2 + x3 * x3;
#pragma unroll
    for (int m = 1; m < 32; m <<= 1) ss += __shfl_xor(ss, m, 32);
    const float rinv = rsqrtf(ss * (1.0f / (float)D_) + 1e-6f);
    x0 = x0 * rinv * nw[lane];
    x1 = x1 * rinv * nw[lane + 32];
    x2 = x2 * rinv * nw[lane + 64];
    x3 = x3 * rinv * nw[lane + 96];
    const float* cs = cosT + (size_t)s * D_;
    const float* sn = sinT + (size_t)s * D_;
    float o0 = x0 * cs[lane]      - x2 * sn[lane];        // d < 64: -x[d+64]*sin
    float o1 = x1 * cs[lane + 32] - x3 * sn[lane + 32];
    float o2 = x2 * cs[lane + 64] + x0 * sn[lane + 64];   // d >= 64: +x[d-64]*sin
    float o3 = x3 * cs[lane + 96] + x1 * sn[lane + 96];
    x0 = o0; x1 = o1; x2 = o2; x3 = o3;
  }

  unsigned short* o = out + (((size_t)b * heads + head) * S_ + s) * D_;
  o[lane]      = f32_to_bf16_u(x0);
  o[lane + 32] = f32_to_bf16_u(x1);
  o[lane + 64] = f32_to_bf16_u(x2);
  o[lane + 96] = f32_to_bf16_u(x3);
}

// ---------------------------------------------------------------------------
// Flash-style causal GQA attention, all matmuls via WMMA bf16.
// Block = 8 waves, 128 q-rows per block, wave owns 16 q-rows.
// K tile (32 keys x 128 d, bf16) is fetched by the Tensor Data Mover when
// available: 2D D# with data_size=2B, tile 128x32, LDS padding 4 DWORDs per
// 64-DWORD row (-> padded stride 136 elements). Falls back to async-to-LDS,
// then to the register path. V is register-transposed into [d][key].
// ctx layout: [B, S, NH*D] bf16 (ready for the output GEMM).
// ---------------------------------------------------------------------------
__global__ __launch_bounds__(256) void attn_kernel(
    const unsigned short* __restrict__ Q, const unsigned short* __restrict__ Kh,
    const unsigned short* __restrict__ Vh, unsigned short* __restrict__ ctx) {
  __shared__ unsigned short Ks[32 * 136];   // [key][d], padded
  __shared__ unsigned short Vs[128 * 40];   // [d][key] (transposed), padded
  __shared__ unsigned short Ps[8 * 16 * 32];// per-wave P staging [16 q][32 key]
  const int tid  = threadIdx.x;
  const int lane = tid & 31;
  const int wv   = tid >> 5;
  const int half = lane >> 4;
  const int nn   = lane & 15;
  const int qb = blockIdx.x * 128;
  const int h  = blockIdx.y;
  const int b  = blockIdx.z;
  const int kh = h / GRP_;

  const unsigned short* Qb = Q + (((size_t)b * NH_ + h) * S_ + qb + wv * 16) * D_;
  const unsigned short* Kb = Kh + (((size_t)b * NKV_ + kh) * S_) * D_;
  const unsigned short* Vb = Vh + (((size_t)b * NKV_ + kh) * S_) * D_;

  // Q fragments (16 rows x 128 d) kept in registers for the whole pass
  v16bf qf[4];
#pragma unroll
  for (int c = 0; c < 4; ++c) qf[c] = frag_ld_a(Qb + c * 32, nn, D_, lane);

  v8f zero = {};
  v8f acc[8];
#pragma unroll
  for (int t = 0; t < 8; ++t) acc[t] = zero;
  float m_i[8], l_i[8];
#pragma unroll
  for (int r = 0; r < 8; ++r) { m_i[r] = -1e30f; l_i[r] = 0.0f; }

  const int qmax = qb + wv * 16 + 15;
  const int ntiles = qb / 32 + 4;  // covers causal extent of last wave in block
  unsigned short* Pw = Ps + wv * 512;

  for (int j = 0; j < ntiles; ++j) {
    const int kb0 = j * 32;

    // ---- K tile -> Ks -------------------------------------------------
#if USE_TDM
    if (wv == 0) {  // one TDM descriptor per block, wave-uniform operands
      const size_t ga = (size_t)(const void*)(Kb + (size_t)kb0 * D_);
      u32x4 g0;
      g0[0] = 1u;                                        // count=1 (valid D#)
      g0[1] = lds_offset(Ks);                            // LDS byte address
      g0[2] = (unsigned)ga;                              // global_addr lo
      g0[3] = ((unsigned)(ga >> 32) & 0x01ffffffu) |     // global_addr hi
              0x80000000u;                               // type=2 ("image")
      i32x8 g1;
      g1[0] = (int)((1u << 16)      // data_size = 2 bytes
                  | (1u << 20)      // pad_enable
                  | (5u << 22)      // pad_interval: every 64 DWORDs (one row)
                  | (3u << 25));    // pad_amount: 4 DWORDs (16B -> stride 136)
      g1[1] = (int)(128u << 16);    // tensor_dim0 = 128 (elements per row)
      g1[2] = (int)(32u << 16);     // tensor_dim1 = 32 (rows from tile start)
      g1[3] = (int)(128u << 16);    // tile_dim0 = 128
      g1[4] = 32;                   // tile_dim1 = 32, tile_dim2 = 0
      g1[5] = (int)(size_t)D_;      // tensor_dim0_stride = 128 elements
      g1[6] = 0;
      g1[7] = 0;
      i32x4 gz4 = {};               // groups 2/3 unused for 2D tile
      i32x8 gz8 = {};
      __builtin_amdgcn_tensor_load_to_lds(g0, g1, gz4, gz4, gz8, 0);
    }
#elif USE_ASYNC_LDS
#pragma unroll
    for (int i = 0; i < 2; ++i) {
      int idx = tid + i * 256;
      int key = idx >> 4;
      int cc  = idx & 15;
      async_b128(Kb + (size_t)(kb0 + key) * D_ + cc * 8,
                 &Ks[key * 136 + cc * 8]);
    }
#else
#pragma unroll
    for (int i = 0; i < 2; ++i) {
      int idx = tid + i * 256;
      int key = idx >> 4;
      int cc  = idx & 15;
      uint4 kk = *(const uint4*)(Kb + (size_t)(kb0 + key) * D_ + cc * 8);
      *(uint4*)(&Ks[key * 136 + cc * 8]) = kk;
    }
#endif

    // ---- V tile -> Vs (transposed, register path) ----------------------
#pragma unroll
    for (int i = 0; i < 2; ++i) {
      int idx = tid + i * 256;
      int key = idx >> 4;
      int cc  = idx & 15;
      uint4 vvv = *(const uint4*)(Vb + (size_t)(kb0 + key) * D_ + cc * 8);
      const unsigned short* pv = (const unsigned short*)&vvv;
#pragma unroll
      for (int e = 0; e < 8; ++e) Vs[(cc * 8 + e) * 40 + key] = pv[e];
    }

#if USE_TDM
    if (wv == 0) __builtin_amdgcn_s_wait_tensorcnt(0);
#elif USE_ASYNC_LDS
    __builtin_amdgcn_s_wait_asynccnt(0);
#endif
    __syncthreads();

    if (kb0 <= qmax) {  // wave-uniform: EXEC stays all-ones inside
      // scores: Q(16x128) @ K^T(128x32) -> two 16x16 fp32 tiles
      v8f s0 = zero, s1 = zero;
#pragma unroll
      for (int c = 0; c < 4; ++c) {
        s0 = __builtin_amdgcn_wmma_f32_16x16x32_bf16(
            false, qf[c], false, frag_ld_b(Ks + c * 32, nn, 136, lane),
            (short)0, s0, false, false);
        s1 = __builtin_amdgcn_wmma_f32_16x16x32_bf16(
            false, qf[c], false, frag_ld_b(Ks + c * 32, 16 + nn, 136, lane),
            (short)0, s1, false, false);
      }
      float alpha[8];
#pragma unroll
      for (int r = 0; r < 8; ++r) {
        const int qg = qb + wv * 16 + r + 8 * half;
        float a  = s0[r] * SCALE_ - ((kb0 + nn)      > qg ? 10000.0f : 0.0f);
        float bb = s1[r] * SCALE_ - ((kb0 + 16 + nn) > qg ? 10000.0f : 0.0f);
        float t = fmaxf(a, bb);                 // row reduce within 16 lanes
        t = fmaxf(t, __shfl_xor(t, 1, 32));
        t = fmaxf(t, __shfl_xor(t, 2, 32));
        t = fmaxf(t, __shfl_xor(t, 4, 32));
        t = fmaxf(t, __shfl_xor(t, 8, 32));
        const float mn = fmaxf(m_i[r], t);
        alpha[r] = __expf(m_i[r] - mn);
        m_i[r] = mn;
        const float p0 = __expf(a - mn);
        const float p1 = __expf(bb - mn);
        const int m = r + 8 * half;
        Pw[m * 32 + nn]      = f32_to_bf16_u(p0);
        Pw[m * 32 + 16 + nn] = f32_to_bf16_u(p1);
        float ps = p0 + p1;
        ps += __shfl_xor(ps, 1, 32);
        ps += __shfl_xor(ps, 2, 32);
        ps += __shfl_xor(ps, 4, 32);
        ps += __shfl_xor(ps, 8, 32);
        l_i[r] = l_i[r] * alpha[r] + ps;
      }
#pragma unroll
      for (int t = 0; t < 8; ++t)
#pragma unroll
        for (int r = 0; r < 8; ++r) acc[t][r] = acc[t][r] * alpha[r];

      // P (16x32) @ V (32x128): 8 WMMAs into output accumulators
      const v16bf pa = frag_ld_a(Pw, nn, 32, lane);
#pragma unroll
      for (int t = 0; t < 8; ++t)
        acc[t] = __builtin_amdgcn_wmma_f32_16x16x32_bf16(
            false, pa, false, frag_ld_b(Vs, t * 16 + nn, 40, lane),
            (short)0, acc[t], false, false);
    }
    __syncthreads();
  }

#pragma unroll
  for (int r = 0; r < 8; ++r) l_i[r] = 1.0f / l_i[r];
#pragma unroll
  for (int t = 0; t < 8; ++t)
#pragma unroll
    for (int r = 0; r < 8; ++r) {
      const int qg = qb + wv * 16 + r + 8 * half;
      ctx[((size_t)b * S_ + qg) * (NH_ * D_) + (size_t)h * D_ + t * 16 + nn] =
          f32_to_bf16_u(acc[t][r] * l_i[r]);
    }
}

// ---------------------------------------------------------------------------
// Host-side orchestration
// ---------------------------------------------------------------------------
extern "C" void kernel_launch(void* const* d_in, const int* in_sizes, int n_in,
                              void* d_out, int out_size, void* d_ws,
                              size_t ws_size, hipStream_t stream) {
  (void)in_sizes; (void)n_in; (void)out_size; (void)ws_size;
  const float* hs   = (const float*)d_in[0];
  const float* cosT = (const float*)d_in[1];
  const float* sinT = (const float*)d_in[2];
  const float* wq   = (const float*)d_in[3];
  const float* wk   = (const float*)d_in[4];
  const float* wv   = (const float*)d_in[5];
  const float* wo   = (const float*)d_in[6];
  const float* qnw  = (const float*)d_in[7];
  const float* knw  = (const float*)d_in[8];

  char* ws = (char*)d_ws;
  size_t off = 0;
  auto carve = [&](size_t bytes) -> void* {
    void* p = ws + off;
    off += (bytes + 255) & ~(size_t)255;
    return p;
  };

  const size_t nTok = (size_t)B_ * S_;       // 4096
  const size_t nHS  = nTok * HID_;           // 8.39M
  const size_t nQ   = nTok * NH_ * D_;       // 16.78M
  const size_t nKV  = nTok * NKV_ * D_;      // 4.19M

  unsigned short* hsb = (unsigned short*)carve(nHS * 2);
  unsigned short* wqb = (unsigned short*)carve((size_t)NH_ * D_ * HID_ * 2);
  unsigned short* wkb = (unsigned short*)carve((size_t)NKV_ * D_ * HID_ * 2);
  unsigned short* wvb = (unsigned short*)carve((size_t)NKV_ * D_ * HID_ * 2);
  unsigned short* wob = (unsigned short*)carve((size_t)HID_ * NH_ * D_ * 2);
  float* qf = (float*)carve(nQ * 4);
  float* kf = (float*)carve(nKV * 4);
  float* vf = (float*)carve(nKV * 4);
  unsigned short* qr  = (unsigned short*)carve(nQ * 2);
  unsigned short* kr  = (unsigned short*)carve(nKV * 2);
  unsigned short* vr  = (unsigned short*)carve(nKV * 2);
  unsigned short* ctx = (unsigned short*)carve(nQ * 2);

  auto cvt = [&](const float* src, unsigned short* dst, size_t n) {
    cvt_bf16_kernel<<<dim3((unsigned)(n / 1024)), dim3(256), 0, stream>>>(
        src, dst, (int)n);
  };
  cvt(hs, hsb, nHS);
  cvt(wq, wqb, (size_t)NH_ * D_ * HID_);
  cvt(wk, wkb, (size_t)NKV_ * D_ * HID_);
  cvt(wv, wvb, (size_t)NKV_ * D_ * HID_);
  cvt(wo, wob, (size_t)HID_ * NH_ * D_);

  // QKV projections: [4096, 2048] x [N, 2048]^T
  gemm_bf16_kernel<<<dim3(NH_ * D_ / 128, (unsigned)(nTok / 128)), 256, 0,
                     stream>>>(hsb, wqb, qf, (int)nTok, NH_ * D_, HID_, 0);
  gemm_bf16_kernel<<<dim3(NKV_ * D_ / 128, (unsigned)(nTok / 128)), 256, 0,
                     stream>>>(hsb, wkb, kf, (int)nTok, NKV_ * D_, HID_, 0);
  gemm_bf16_kernel<<<dim3(NKV_ * D_ / 128, (unsigned)(nTok / 128)), 256, 0,
                     stream>>>(hsb, wvb, vf, (int)nTok, NKV_ * D_, HID_, 0);

  // RMSNorm + RoPE + relayout to [B, H, S, D] bf16 (V: relayout only)
  normrope_kernel<<<dim3((unsigned)(nTok * NH_ / 8)), 256, 0, stream>>>(
      qf, qnw, cosT, sinT, qr, NH_, 1);
  normrope_kernel<<<dim3((unsigned)(nTok * NKV_ / 8)), 256, 0, stream>>>(
      kf, knw, cosT, sinT, kr, NKV_, 1);
  normrope_kernel<<<dim3((unsigned)(nTok * NKV_ / 8)), 256, 0, stream>>>(
      vf, knw, cosT, sinT, vr, NKV_, 0);

  // Flash attention
  attn_kernel<<<dim3(S_ / 128, NH_, B_), 256, 0, stream>>>(qr, kr, vr, ctx);

  // Output projection: [4096, 4096] x [2048, 4096]^T -> fp32 d_out
  gemm_bf16_kernel<<<dim3(HID_ / 128, (unsigned)(nTok / 128)), 256, 0,
                     stream>>>(ctx, wob, d_out, (int)nTok, HID_, NH_ * D_, 0);
}
